// QueryTokenAggregator_87557203296538
// MI455X (gfx1250) — matrix operand order repaired and checked by reading, not compiled, not run on hardware
//
#include <hip/hip_runtime.h>
#include <hip/hip_bf16.h>

// ---------------- problem constants (from reference) ----------------
constexpr int B_   = 8;
constexpr int Din_ = 512;
constexpr int T_   = 1024;
constexpr int G_   = 128;
constexpr int DM_  = 512;
constexpr int H_   = 8;
constexpr int L_   = 2;
constexpr int FF_  = 2048;
constexpr int S_   = T_ + G_;      // 1152 tokens
constexpr int DH_  = DM_ / H_;     // 64
constexpr float SCALE_ = 0.125f;   // 1/sqrt(64)

typedef _Float16 h16;
typedef __attribute__((ext_vector_type(16))) _Float16 v16h;
typedef __attribute__((ext_vector_type(8)))  _Float16 v8h;
typedef __attribute__((ext_vector_type(8)))  float    v8f;

#define WMMA_F16(a, b, c) \
  __builtin_amdgcn_wmma_f32_16x16x32_f16(false, (a), false, (b), (short)0, (c), false, false)

// ---- CDNA5 async load-to-LDS path (ASYNCcnt), with sync fallback ----
#if defined(__has_builtin)
#if __has_builtin(__builtin_amdgcn_global_load_async_to_lds_b128)
#define GEMM_ASYNC 1
#endif
#endif
#ifndef GEMM_ASYNC
#define GEMM_ASYNC 0
#endif

#if GEMM_ASYNC
typedef int v4i __attribute__((vector_size(16)));
typedef __attribute__((address_space(1))) v4i* gp128_t;
typedef __attribute__((address_space(3))) v4i* lp128_t;
__device__ __forceinline__ void async_b128(const void* g, void* l) {
  __builtin_amdgcn_global_load_async_to_lds_b128((gp128_t)g, (lp128_t)l, 0, 0);
}
__device__ __forceinline__ void wait_async0() {
  asm volatile("s_wait_asynccnt 0x0" ::: "memory");
}
#endif

__device__ __forceinline__ float gelu_tanh(float x) {
  return 0.5f * x * (1.0f + tanhf(0.7978845608028654f * (x + 0.044715f * x * x * x)));
}

// ---------------- prep: group stats + interleave permutation ----------------
__global__ __launch_bounds__(256) void prep_kernel(
    const float* __restrict__ align, const int* __restrict__ nseg,
    int* __restrict__ glast, float* __restrict__ invcnt,
    int* __restrict__ idx, int* __restrict__ qpos) {
  int b = blockIdx.x;
  int tid = threadIdx.x;
  __shared__ int nb[T_];
  __shared__ int sFL;
  const float* ab = align + (long)b * G_ * T_;
  int n = nseg[b];

  for (int g = tid; g < G_; g += 256) {
    int last = 0; float cnt = 0.0f;
    const float* ag = ab + (long)g * T_;
    for (int t = 0; t < T_; ++t) {
      float a = ag[t];
      if (a != 0.0f) { cnt += a; last = t; }
    }
    glast[b * G_ + g]  = last;
    invcnt[b * G_ + g] = 1.0f / fmaxf(cnt, 1.0f);
  }
  for (int t = tid; t < T_; t += 256) nb[t] = 0;
  __syncthreads();
  if (tid == 0) {
    int fl = 0;
    for (int g = 0; g < n; ++g) fl = max(fl, glast[b * G_ + g] + 1);
    sFL = fl;
  }
  __syncthreads();
  for (int g = tid; g < G_; g += 256) {
    if (g < n) {
      int pos = glast[b * G_ + g] + 1;
      if (pos < T_) atomicAdd(&nb[pos], 1);
    }
  }
  __syncthreads();
  if (tid == 0) {  // inclusive prefix: nb[t] = #valid groups with last < t
    int run = 0;
    for (int t = 0; t < T_; ++t) { run += nb[t]; nb[t] = run; }
  }
  __syncthreads();
  int FL = sFL, VC = FL + n;
  for (int t = tid; t < T_; t += 256) {
    int dest = (t < FL) ? (t + nb[t]) : (VC + (t - FL));
    idx[b * S_ + dest] = t;
  }
  for (int g = tid; g < G_; g += 256) {
    int dest = (g < n) ? (glast[b * G_ + g] + g + 1)
                       : (VC + (T_ - FL) + (g - n));
    idx[b * S_ + dest] = T_ + g;
    qpos[b * G_ + g]   = dest;
  }
}

// ---------------- features (B,Din,T) -> (B,T,Din) f32 ----------------
__global__ __launch_bounds__(256) void transpose_kernel(
    const float* __restrict__ f, float* __restrict__ ft) {
  __shared__ float tile[32][33];
  int t0 = blockIdx.x * 32, d0 = blockIdx.y * 32, b = blockIdx.z;
  const float* fb = f + (long)b * Din_ * T_;
  for (int r = threadIdx.y; r < 32; r += 8)
    tile[r][threadIdx.x] = fb[(long)(d0 + r) * T_ + t0 + threadIdx.x];
  __syncthreads();
  float* ftb = ft + (long)b * T_ * Din_;
  for (int r = threadIdx.y; r < 32; r += 8)
    ftb[(long)(t0 + r) * Din_ + d0 + threadIdx.x] = tile[threadIdx.x][r];
}

// ---------------- weight convert+transpose: W (K,N) f32 -> Wt (N,K) f16 ----------------
__global__ __launch_bounds__(256) void wconv_kernel(
    const float* __restrict__ W, h16* __restrict__ Wt, int K, int N) {
  __shared__ float tile[32][33];
  int k0 = blockIdx.x * 32, n0 = blockIdx.y * 32;
  for (int r = threadIdx.y; r < 32; r += 8)
    tile[r][threadIdx.x] = W[(long)(k0 + r) * N + n0 + threadIdx.x];
  __syncthreads();
  for (int r = threadIdx.y; r < 32; r += 8)
    Wt[(long)(n0 + r) * K + k0 + threadIdx.x] = (h16)tile[threadIdx.x][r];
}

// ---------------- segment-mean queries (B,G,Din) f32 ----------------
__global__ __launch_bounds__(128) void queries_kernel(
    const float* __restrict__ ft, const int* __restrict__ glast,
    const float* __restrict__ invcnt, const int* __restrict__ nseg,
    float* __restrict__ qout) {
  int g = blockIdx.x, b = blockIdx.y, tid = threadIdx.x;
  float* qg = qout + ((long)b * G_ + g) * Din_;
  int n = nseg[b];
  if (g >= n) {
    for (int d = tid; d < Din_; d += 128) qg[d] = 0.0f;
    return;
  }
  int lo = (g == 0) ? 0 : glast[b * G_ + g - 1] + 1;
  int hi = glast[b * G_ + g];
  float ic = invcnt[b * G_ + g];
  const float* fb = ft + (long)b * T_ * Din_;
  for (int d = tid; d < Din_; d += 128) {
    float s = 0.0f;
    for (int t = lo; t <= hi; ++t) s += fb[(long)t * Din_ + d];
    qg[d] = s * ic;
  }
}

// ---------------- interleave gather -> xt (B,S,Din) f16 ----------------
__global__ __launch_bounds__(128) void gather_kernel(
    const float* __restrict__ ft, const float* __restrict__ qrs,
    const int* __restrict__ idx, h16* __restrict__ xt) {
  int s = blockIdx.x, b = blockIdx.y, tid = threadIdx.x;
  int j = idx[b * S_ + s];
  const float* src = (j < T_) ? (ft  + ((long)b * T_ + j)        * Din_)
                              : (qrs + ((long)b * G_ + (j - T_)) * Din_);
  h16* dst = xt + ((long)b * S_ + s) * Din_;
  for (int d = tid; d < Din_; d += 128) dst[d] = (h16)src[d];
}

// ---------------- layer norm: h f32 -> y f16 ----------------
__global__ __launch_bounds__(128) void ln_kernel(
    const float* __restrict__ x, h16* __restrict__ y,
    const float* __restrict__ sc, const float* __restrict__ bi) {
  long row = (long)blockIdx.y * S_ + blockIdx.x;
  const float* xr = x + row * DM_;
  h16* yr = y + row * DM_;
  int tid = threadIdx.x, lane = tid & 31, wid = tid >> 5;
  float sum = 0.0f, sq = 0.0f;
  for (int i = tid; i < DM_; i += 128) {
    float v = xr[i]; sum += v; sq += v * v;
  }
  for (int off = 16; off >= 1; off >>= 1) {
    sum += __shfl_xor(sum, off, 32);
    sq  += __shfl_xor(sq,  off, 32);
  }
  __shared__ float red[8];
  if (lane == 0) { red[wid] = sum; red[4 + wid] = sq; }
  __syncthreads();
  if (tid == 0) {
    float a = 0.0f, c = 0.0f;
    for (int w = 0; w < 4; ++w) { a += red[w]; c += red[4 + w]; }
    red[0] = a; red[4] = c;
  }
  __syncthreads();
  float mean = red[0] * (1.0f / DM_);
  float var  = red[4] * (1.0f / DM_) - mean * mean;
  float inv  = rsqrtf(var + 1e-5f);
  for (int i = tid; i < DM_; i += 128)
    yr[i] = (h16)((xr[i] - mean) * inv * sc[i] + bi[i]);
}

// ---------------- RoPE on f16 q and k ----------------
__global__ __launch_bounds__(256) void rope_kernel(
    h16* __restrict__ q, h16* __restrict__ k) {
  int s = blockIdx.x, b = blockIdx.y;
  int tid = threadIdx.x;
  int hh = tid >> 5;          // head 0..7
  int i  = tid & 31;          // dim in half 0..31
  float fr  = __expf(-(float)i * (9.210340371976184f / 32.0f));  // 1/10000^(i/32)
  float ang = (float)s * fr;
  float sn, cs;
  __sincosf(ang, &sn, &cs);
  long off = ((long)b * S_ + s) * DM_ + hh * DH_;
  float x1 = (float)q[off + i], x2 = (float)q[off + 32 + i];
  q[off + i]      = (h16)(x1 * cs - x2 * sn);
  q[off + 32 + i] = (h16)(x1 * sn + x2 * cs);
  x1 = (float)k[off + i]; x2 = (float)k[off + 32 + i];
  k[off + i]      = (h16)(x1 * cs - x2 * sn);
  k[off + 32 + i] = (h16)(x1 * sn + x2 * cs);
}

// ---------------- WMMA GEMM: epi(A_f16 (B,M,K) @ Wt_f16 (N,K)^T) ----------------
// Double-buffered LDS pipeline; async DMA to LDS when available.
// out_mode: 0 = f32 (M,N) [+resid/ls], 1 = f16 (M,N), 2 = f16 transposed (N,M)
__global__ __launch_bounds__(256) void gemm_wmma_kernel(
    const h16* __restrict__ A, const h16* __restrict__ Wt,
    float* __restrict__ Cf, h16* __restrict__ Ch,
    const float* __restrict__ bias, const float* __restrict__ resid,
    const float* __restrict__ ls, int M, int N, int K,
    int gelu_flag, int out_mode, float outscale) {
  int b = blockIdx.z;
  const long strideA = (long)M * K, strideC = (long)M * N;
  A += (long)b * strideA;
  if (Cf) Cf += (long)b * strideC;
  if (Ch) Ch += (long)b * strideC;
  const float* R = resid ? resid + (long)b * strideC : nullptr;

  __shared__ h16 As[2][64][40];    // [buf][m][k] padded rows (80 B)
  __shared__ h16 Bs[2][128][40];   // [buf][n][k]

  int m0 = blockIdx.y * 64, n0 = blockIdx.x * 128;
  int tid  = threadIdx.x;
  int lane = tid & 31, wid = tid >> 5;
  int wr = wid >> 1, wc = wid & 1;      // 4x2 waves, each owns 16 rows x 64 cols
  int hi = lane >> 4, lo = lane & 15;

  // staging coordinates: one b128 of A, two b128 of B per thread per tile
  int rA = tid >> 2, c8 = (tid & 3) * 8;
  int rB1 = rA + 64;

  v8f acc[4];
  #pragma unroll
  for (int j = 0; j < 4; ++j)
    #pragma unroll
    for (int i = 0; i < 8; ++i) acc[j][i] = 0.0f;

  // prologue: stage tile 0 into buffer 0
#if GEMM_ASYNC
  async_b128(A  + (long)(m0 + rA)  * K + c8, &As[0][rA][c8]);
  async_b128(Wt + (long)(n0 + rA)  * K + c8, &Bs[0][rA][c8]);
  async_b128(Wt + (long)(n0 + rB1) * K + c8, &Bs[0][rB1][c8]);
  wait_async0();
#else
  *(v8h*)&As[0][rA][c8]  = *(const v8h*)(A  + (long)(m0 + rA)  * K + c8);
  *(v8h*)&Bs[0][rA][c8]  = *(const v8h*)(Wt + (long)(n0 + rA)  * K + c8);
  *(v8h*)&Bs[0][rB1][c8] = *(const v8h*)(Wt + (long)(n0 + rB1) * K + c8);
#endif
  __syncthreads();

  int buf = 0;
  for (int k0 = 0; k0 < K; k0 += 32, buf ^= 1) {
    const int kn = k0 + 32;
    const bool haveNext = kn < K;
#if GEMM_ASYNC
    if (haveNext) {   // DMA next tile into the other buffer while we compute
      async_b128(A  + (long)(m0 + rA)  * K + kn + c8, &As[buf ^ 1][rA][c8]);
      async_b128(Wt + (long)(n0 + rA)  * K + kn + c8, &Bs[buf ^ 1][rA][c8]);
      async_b128(Wt + (long)(n0 + rB1) * K + kn + c8, &Bs[buf ^ 1][rB1][c8]);
    }
#else
    v8h nA{}, nB0{}, nB1{};
    if (haveNext) {   // fetch next tile into registers while we compute
      nA  = *(const v8h*)(A  + (long)(m0 + rA)  * K + kn + c8);
      nB0 = *(const v8h*)(Wt + (long)(n0 + rA)  * K + kn + c8);
      nB1 = *(const v8h*)(Wt + (long)(n0 + rB1) * K + kn + c8);
    }
#endif

    // A fragment (16x32), reused by 4 WMMAs
    int am = wr * 16 + lo;
    int kh = hi * 8;
    v8h a0 = *(const v8h*)&As[buf][am][kh];
    v8h a1 = *(const v8h*)&As[buf][am][16 + kh];
    v16h af;
    #pragma unroll
    for (int i = 0; i < 8; ++i) { af[i] = a0[i]; af[8 + i] = a1[i]; }

    #pragma unroll
    for (int j = 0; j < 4; ++j) {
      int bn = wc * 64 + j * 16 + lo;
      int kb = hi * 8;
      v8h b0 = *(const v8h*)&Bs[buf][bn][kb];
      v8h b1 = *(const v8h*)&Bs[buf][bn][kb + 16];
      v16h bf;
      #pragma unroll
      for (int i = 0; i < 8; ++i) { bf[i] = b0[i]; bf[8 + i] = b1[i]; }
      acc[j] = WMMA_F16(af, bf, acc[j]);
    }

#if GEMM_ASYNC
    if (haveNext) wait_async0();
#else
    if (haveNext) {
      *(v8h*)&As[buf ^ 1][rA][c8]  = nA;
      *(v8h*)&Bs[buf ^ 1][rA][c8]  = nB0;
      *(v8h*)&Bs[buf ^ 1][rB1][c8] = nB1;
    }
#endif
    __syncthreads();
  }

  // epilogue (C layout: row = i + 8*hi, col = lo within each 16x16 tile)
  #pragma unroll
  for (int j = 0; j < 4; ++j) {
    int gn = n0 + wc * 64 + j * 16 + lo;
    float bval = bias ? bias[gn] : 0.0f;
    float lsv  = ls ? ls[gn] : 1.0f;
    if (out_mode == 2) {
      // f16 transposed store: 8 consecutive m per lane -> one 16B store
      v8h pack;
      #pragma unroll
      for (int i = 0; i < 8; ++i) {
        float v = acc[j][i] + bval;
        if (gelu_flag) v = gelu_tanh(v);
        pack[i] = (h16)(v * outscale);
      }
      int gm0 = m0 + wr * 16 + hi * 8;
      *(v8h*)(Ch + (long)gn * M + gm0) = pack;
    } else if (out_mode == 1) {
      #pragma unroll
      for (int i = 0; i < 8; ++i) {
        int gm = m0 + wr * 16 + hi * 8 + i;
        float v = acc[j][i] + bval;
        if (gelu_flag) v = gelu_tanh(v);
        Ch[(long)gm * N + gn] = (h16)(v * outscale);
      }
    } else {
      #pragma unroll
      for (int i = 0; i < 8; ++i) {
        int gm = m0 + wr * 16 + hi * 8 + i;
        float v = acc[j][i] + bval;
        if (gelu_flag) v = gelu_tanh(v);
        v *= outscale;
        if (R) v = R[(long)gm * N + gn] + lsv * v;
        Cf[(long)gm * N + gn] = v;
      }
    }
  }
}

// ---------------- flash attention: one wave per 16-query tile ----------------
// q,k: (B,S,DM) f16 (q pre-scaled); vT: (B,DM,S) f16; o: (B,S,DM) f16
__global__ __launch_bounds__(32) void flash_attn_kernel(
    const h16* __restrict__ q, const h16* __restrict__ k,
    const h16* __restrict__ vT, h16* __restrict__ o) {
  int qt = blockIdx.x, hh = blockIdx.y, b = blockIdx.z;
  int lane = threadIdx.x;
  int lo = lane & 15, hi = lane >> 4;
  const long base = (long)b * S_ * DM_;
  const int col0 = hh * DH_;

  // Q fragments (2 frags cover dh=64), direct f16 vector loads
  v16h qa[2];
  {
    const h16* qp = q + base + (long)(qt * 16 + lo) * DM_ + col0;
    #pragma unroll
    for (int f = 0; f < 2; ++f) {
      int d0 = f * 32 + hi * 8;
      v8h q0 = *(const v8h*)(qp + d0);
      v8h q1 = *(const v8h*)(qp + d0 + 16);
      #pragma unroll
      for (int i = 0; i < 8; ++i) { qa[f][i] = q0[i]; qa[f][8 + i] = q1[i]; }
    }
  }

  float mrow[8], lrow[8];
  v8f oacc[4];
  #pragma unroll
  for (int i = 0; i < 8; ++i) { mrow[i] = -1e30f; lrow[i] = 0.0f; }
  #pragma unroll
  for (int f = 0; f < 4; ++f)
    #pragma unroll
    for (int i = 0; i < 8; ++i) oacc[f][i] = 0.0f;

  __shared__ h16 P[16][40];   // probs tile (16 q x 32 k) relayout buffer

  for (int kt = 0; kt < S_ / 32; ++kt) {
    int kb0 = kt * 32;
    if (kt + 1 < S_ / 32) {   // prefetch next K/V tiles
      __builtin_prefetch(k + base + (long)(kb0 + 32 + lo) * DM_ + col0, 0, 1);
      __builtin_prefetch(vT + ((long)b * DM_ + col0 + lo) * S_ + kb0 + 32, 0, 1);
    }
    v8f sc[2];
    #pragma unroll
    for (int khf = 0; khf < 2; ++khf) {
      #pragma unroll
      for (int i = 0; i < 8; ++i) sc[khf][i] = 0.0f;
      const h16* kp = k + base + (long)(kb0 + khf * 16 + lo) * DM_ + col0;
      #pragma unroll
      for (int f = 0; f < 2; ++f) {
        int d0 = f * 32 + hi * 8;
        v8h k0v = *(const v8h*)(kp + d0);
        v8h k1v = *(const v8h*)(kp + d0 + 16);
        v16h bf;
        #pragma unroll
        for (int i = 0; i < 8; ++i) { bf[i] = k0v[i]; bf[8 + i] = k1v[i]; }
        sc[khf] = WMMA_F16(qa[f], bf, sc[khf]);
      }
    }
    // online softmax per row (rows i+8*hi, cols spread over 16-lane group)
    float alpha[8];
    #pragma unroll
    for (int i = 0; i < 8; ++i) {
      float tm = fmaxf(sc[0][i], sc[1][i]);
      for (int off = 8; off >= 1; off >>= 1) tm = fmaxf(tm, __shfl_xor(tm, off, 32));
      float mn = fmaxf(mrow[i], tm);
      alpha[i] = __expf(mrow[i] - mn);
      mrow[i] = mn;
      float p0 = __expf(sc[0][i] - mn);
      float p1 = __expf(sc[1][i] - mn);
      float ts = p0 + p1;
      for (int off = 8; off >= 1; off >>= 1) ts += __shfl_xor(ts, off, 32);
      lrow[i] = lrow[i] * alpha[i] + ts;
      P[hi * 8 + i][lo]      = (h16)p0;
      P[hi * 8 + i][16 + lo] = (h16)p1;
    }
    #pragma unroll
    for (int f = 0; f < 4; ++f)
      #pragma unroll
      for (int i = 0; i < 8; ++i) oacc[f][i] *= alpha[i];
    __syncthreads();
    // P as A fragment (K = key index 0..31)
    v16h pa;
    {
      int kh = hi * 8;
      #pragma unroll
      for (int i = 0; i < 8; ++i) {
        pa[i]     = P[lo][kh + i];
        pa[8 + i] = P[lo][16 + kh + i];
      }
    }
    __syncthreads();
    // P @ V via vT: contiguous key runs per lane
    #pragma unroll
    for (int f = 0; f < 4; ++f) {
      const h16* vp = vT + ((long)b * DM_ + col0 + f * 16 + lo) * S_ + kb0 + hi * 8;
      v8h v0 = *(const v8h*)vp;
      v8h v1 = *(const v8h*)(vp + 16);
      v16h vf;
      #pragma unroll
      for (int i = 0; i < 8; ++i) { vf[i] = v0[i]; vf[8 + i] = v1[i]; }
      oacc[f] = WMMA_F16(pa, vf, oacc[f]);
    }
  }

  h16* op = o + base + (long)(qt * 16) * DM_ + col0;
  #pragma unroll
  for (int f = 0; f < 4; ++f)
    #pragma unroll
    for (int i = 0; i < 8; ++i) {
      int r = hi * 8 + i;
      op[(long)r * DM_ + f * 16 + lo] = (h16)(oacc[f][i] / lrow[i]);
    }
}

// ---------------- output gather: (B,Din,G) f32 ----------------
__global__ __launch_bounds__(128) void output_kernel(
    const float* __restrict__ tout, const int* __restrict__ qpos,
    const int* __restrict__ nseg, float* __restrict__ out) {
  int d = blockIdx.x, b = blockIdx.y, g = threadIdx.x;
  int n = nseg[b];
  float v = 0.0f;
  if (g < n) {
    int p = qpos[b * G_ + g];
    v = tout[((long)b * S_ + p) * DM_ + d];
  }
  out[((long)b * Din_ + d) * G_ + g] = v;
}

// ---------------- host-side orchestration ----------------
static inline void launch_gemm(const h16* A, const h16* Wt, float* Cf, h16* Ch,
                               const float* bias, const float* resid, const float* ls,
                               int M, int N, int K, int gelu, int mode, float oscale,
                               hipStream_t st) {
  dim3 grid(N / 128, M / 64, B_);
  gemm_wmma_kernel<<<grid, 256, 0, st>>>(A, Wt, Cf, Ch, bias, resid, ls,
                                         M, N, K, gelu, mode, oscale);
}

static inline void launch_wconv(const float* W, h16* Wt, int K, int N, hipStream_t st) {
  wconv_kernel<<<dim3(K / 32, N / 32), dim3(32, 8), 0, st>>>(W, Wt, K, N);
}

extern "C" void kernel_launch(void* const* d_in, const int* in_sizes, int n_in,
                              void* d_out, int out_size, void* d_ws, size_t ws_size,
                              hipStream_t stream) {
  (void)in_sizes; (void)n_in; (void)out_size; (void)ws_size;
  const float* features = (const float*)d_in[0];
  const float* align    = (const float*)d_in[1];
  const int*   nseg     = (const int*)  d_in[2];
  const float* W_in  = (const float*)d_in[3];
  const float* b_in  = (const float*)d_in[4];
  const float* ln1_s = (const float*)d_in[5];
  const float* ln1_b = (const float*)d_in[6];
  const float* Wq    = (const float*)d_in[7];
  const float* Wk    = (const float*)d_in[8];
  const float* Wv    = (const float*)d_in[9];
  const float* Wo    = (const float*)d_in[10];
  const float* ls1   = (const float*)d_in[11];
  const float* ln2_s = (const float*)d_in[12];
  const float* ln2_b = (const float*)d_in[13];
  const float* W1    = (const float*)d_in[14];
  const float* b1    = (const float*)d_in[15];
  const float* W2    = (const float*)d_in[16];
  const float* b2    = (const float*)d_in[17];
  const float* ls2   = (const float*)d_in[18];
  const float* lnf_s = (const float*)d_in[19];
  const float* lnf_b = (const float*)d_in[20];
  const float* W_out = (const float*)d_in[21];
  const float* b_out = (const float*)d_in[22];

  // workspace partition (256B-aligned slots)
  char* wsb = (char*)d_ws;
  size_t off = 0;
  auto alloc = [&](size_t bytes) {
    void* p = wsb + off;
    off += (bytes + 255) & ~(size_t)255;
    return p;
  };
  float* ft   = (float*)alloc((size_t)B_ * T_ * Din_ * 4);
  float* qrs  = (float*)alloc((size_t)B_ * G_ * Din_ * 4);
  float* hbuf = (float*)alloc((size_t)B_ * S_ * DM_ * 4);
  float* tout = (float*)alloc((size_t)B_ * S_ * DM_ * 4);
  h16* xt  = (h16*)alloc((size_t)B_ * S_ * Din_ * 2);
  h16* y   = (h16*)alloc((size_t)B_ * S_ * DM_ * 2);
  h16* qf  = (h16*)alloc((size_t)B_ * S_ * DM_ * 2);
  h16* kf  = (h16*)alloc((size_t)B_ * S_ * DM_ * 2);
  h16* vt  = (h16*)alloc((size_t)B_ * DM_ * S_ * 2);
  h16* obf = (h16*)alloc((size_t)B_ * S_ * DM_ * 2);
  h16* ffb = (h16*)alloc((size_t)B_ * S_ * FF_ * 2);
  h16* wt_in  = (h16*)alloc((size_t)DM_ * Din_ * 2);
  h16* wt_q   = (h16*)alloc((size_t)L_ * DM_ * DM_ * 2);
  h16* wt_k   = (h16*)alloc((size_t)L_ * DM_ * DM_ * 2);
  h16* wt_v   = (h16*)alloc((size_t)L_ * DM_ * DM_ * 2);
  h16* wt_o   = (h16*)alloc((size_t)L_ * DM_ * DM_ * 2);
  h16* wt_1   = (h16*)alloc((size_t)L_ * FF_ * DM_ * 2);
  h16* wt_2   = (h16*)alloc((size_t)L_ * DM_ * FF_ * 2);
  h16* wt_out = (h16*)alloc((size_t)Din_ * DM_ * 2);
  int*   glast = (int*)  alloc((size_t)B_ * G_ * 4);
  float* invc  = (float*)alloc((size_t)B_ * G_ * 4);
  int*   idx   = (int*)  alloc((size_t)B_ * S_ * 4);
  int*   qpos  = (int*)  alloc((size_t)B_ * G_ * 4);

  // weights -> f16 transposed (N,K)
  launch_wconv(W_in, wt_in, Din_, DM_, stream);
  for (int l = 0; l < L_; ++l) {
    const size_t mm = (size_t)l * DM_ * DM_;
    launch_wconv(Wq + mm, wt_q + mm, DM_, DM_, stream);
    launch_wconv(Wk + mm, wt_k + mm, DM_, DM_, stream);
    launch_wconv(Wv + mm, wt_v + mm, DM_, DM_, stream);
    launch_wconv(Wo + mm, wt_o + mm, DM_, DM_, stream);
    launch_wconv(W1 + (size_t)l * DM_ * FF_, wt_1 + (size_t)l * FF_ * DM_, DM_, FF_, stream);
    launch_wconv(W2 + (size_t)l * FF_ * DM_, wt_2 + (size_t)l * DM_ * FF_, FF_, DM_, stream);
  }
  launch_wconv(W_out, wt_out, DM_, Din_, stream);

  // permutation / segment bookkeeping, transpose, queries, interleave
  prep_kernel<<<B_, 256, 0, stream>>>(align, nseg, glast, invc, idx, qpos);
  transpose_kernel<<<dim3(T_ / 32, Din_ / 32, B_), dim3(32, 8), 0, stream>>>(features, ft);
  queries_kernel<<<dim3(G_, B_), 128, 0, stream>>>(ft, glast, invc, nseg, qrs);
  gather_kernel<<<dim3(S_, B_), 128, 0, stream>>>(ft, qrs, idx, xt);

  // input projection -> h (f32)
  launch_gemm(xt, wt_in, hbuf, nullptr, b_in, nullptr, nullptr,
              S_, DM_, Din_, 0, 0, 1.0f, stream);

  for (int l = 0; l < L_; ++l) {
    const size_t mm = (size_t)l * DM_ * DM_;
    ln_kernel<<<dim3(S_, B_), 128, 0, stream>>>(hbuf, y, ln1_s + l * DM_, ln1_b + l * DM_);
    // q scaled by 1/sqrt(dh) (RoPE is linear, scaling commutes)
    launch_gemm(y, wt_q + mm, nullptr, qf, nullptr, nullptr, nullptr,
                S_, DM_, DM_, 0, 1, SCALE_, stream);
    launch_gemm(y, wt_k + mm, nullptr, kf, nullptr, nullptr, nullptr,
                S_, DM_, DM_, 0, 1, 1.0f, stream);
    // v written transposed (DM, S) for contiguous attention B-fragments
    launch_gemm(y, wt_v + mm, nullptr, vt, nullptr, nullptr, nullptr,
                S_, DM_, DM_, 0, 2, 1.0f, stream);
    rope_kernel<<<dim3(S_, B_), 256, 0, stream>>>(qf, kf);
    flash_attn_kernel<<<dim3(S_ / 16, H_, B_), 32, 0, stream>>>(qf, kf, vt, obf);
    launch_gemm(obf, wt_o + mm, hbuf, nullptr, nullptr, hbuf, ls1 + l * DM_,
                S_, DM_, DM_, 0, 0, 1.0f, stream);
    ln_kernel<<<dim3(S_, B_), 128, 0, stream>>>(hbuf, y, ln2_s + l * DM_, ln2_b + l * DM_);
    launch_gemm(y, wt_1 + (size_t)l * FF_ * DM_, nullptr, ffb, b1 + l * FF_, nullptr, nullptr,
                S_, FF_, DM_, 1, 1, 1.0f, stream);
    launch_gemm(ffb, wt_2 + (size_t)l * DM_ * FF_, hbuf, nullptr, b2 + l * DM_, hbuf,
                ls2 + l * DM_, S_, DM_, FF_, 0, 0, 1.0f, stream);
  }

  ln_kernel<<<dim3(S_, B_), 128, 0, stream>>>(hbuf, y, lnf_s, lnf_b);
  launch_gemm(y, wt_out, tout, nullptr, b_out, nullptr, nullptr,
              S_, Din_, DM_, 0, 0, 1.0f, stream);
  output_kernel<<<dim3(Din_, B_), G_, 0, stream>>>(tout, qpos, nseg, (float*)d_out);
}
